// Trifusion_59906203844722
// MI455X (gfx1250) — matrix operand order repaired and checked by reading, not compiled, not run on hardware
//
#include <hip/hip_runtime.h>
#include <hip/hip_bf16.h>

// ---------------------------------------------------------------------------
// Types for CDNA5 WMMA
// ---------------------------------------------------------------------------
typedef __attribute__((ext_vector_type(16))) __bf16 v16bf;
typedef __attribute__((ext_vector_type(8)))  float  v8f;

#define BM 128
#define BN 128
#define BK 32
#define HIDC 256
// fp32-operand GEMM LDS strides (ushorts)
#define ASTR 36
#define BSTR 132
// bf16-operand GEMM LDS strides (ushorts), 16B-aligned rows for b128 stores
#define ASTR2 40
#define BSTR2 136

// branch-free round-to-nearest-even f32 -> bf16
__device__ __forceinline__ unsigned short bf16r(float f) {
    unsigned u = __builtin_bit_cast(unsigned, f);
    u += 0x7FFFu + ((u >> 16) & 1u);
    return (unsigned short)(u >> 16);
}
// pack two f32 -> two bf16 in one dword
__device__ __forceinline__ unsigned pk_bf16(float f0, float f1) {
    unsigned u0 = __builtin_bit_cast(unsigned, f0);
    unsigned u1 = __builtin_bit_cast(unsigned, f1);
    u0 += 0x7FFFu + ((u0 >> 16) & 1u);
    u1 += 0x7FFFu + ((u1 >> 16) & 1u);
    return (u0 >> 16) | (u1 & 0xFFFF0000u);
}
__device__ __forceinline__ __bf16 bfu(unsigned short u) {
    return __builtin_bit_cast(__bf16, u);
}
__device__ __forceinline__ void scatter8(unsigned short* p, uint4 v, const int stride) {
    p[0 * stride] = (unsigned short)(v.x);
    p[1 * stride] = (unsigned short)(v.x >> 16);
    p[2 * stride] = (unsigned short)(v.y);
    p[3 * stride] = (unsigned short)(v.y >> 16);
    p[4 * stride] = (unsigned short)(v.z);
    p[5 * stride] = (unsigned short)(v.z >> 16);
    p[6 * stride] = (unsigned short)(v.w);
    p[7 * stride] = (unsigned short)(v.w >> 16);
}

// ===========================================================================
// GEMM #1 (preferred): bf16 operands, split-K partials.
//   Cpart[blockIdx.z] [M,256](f32) = opA(A)*opB(B) over K-slice
//   OPA==0: A[m,k]=A[m*lda+k]   OPA==1: A[m,k]=A[k*lda+m]
//   OPB==0: B[k,n]=B[k*ldb+n]   OPB==1: B[k,n]=B[n*ldb+k]
// kSpan = K/S per slice; kBeg = blockIdx.z*kSpan; psz = M*256.
// M%128==0, kSpan%32==0, leading dims %8==0 (true for all call sites).
// 256 threads = 8 waves (4x2); wave tile 32x64 = 2x4 WMMA accumulators.
// ===========================================================================
template <int OPA, int OPB>
__global__ __launch_bounds__(256)
void gemm_wmma_h(float* __restrict__ C, const unsigned short* __restrict__ A,
                 const unsigned short* __restrict__ Bp,
                 int kSpan, int lda, int ldb, int ldc, int psz)
{
    __shared__ unsigned short As[BM][ASTR2];
    __shared__ unsigned short Bs[BK][BSTR2];

    const int t    = threadIdx.x;
    const int lane = t & 31;
    const int w    = t >> 5;
    const int wr   = w >> 1;
    const int wc   = w & 1;
    const int half = lane >> 4;
    const int l16  = lane & 15;

    const int m0   = blockIdx.y * BM;
    const int n0   = blockIdx.x * BN;
    const int kBeg = blockIdx.z * kSpan;

    // ---- staging addresses (k-invariant): 2 octs (8 bf16) per thread per tile
    const unsigned short* aBase;
    uint4*          aStV[2];
    unsigned short* aStT[2];
    if constexpr (OPA == 0) {
        aBase = A + (size_t)(m0 + (t >> 2)) * lda + ((t & 3) << 3) + kBeg;
        #pragma unroll
        for (int e = 0; e < 2; ++e)
            aStV[e] = (uint4*)&As[(t >> 2) + e * 64][(t & 3) << 3];
    } else {
        aBase = A + (size_t)((t >> 4) + kBeg) * lda + m0 + ((t & 15) << 3);
        #pragma unroll
        for (int e = 0; e < 2; ++e)
            aStT[e] = &As[(t & 15) << 3][(t >> 4) + e * 16];
    }
    const unsigned short* bBase;
    uint4*          bStV[2];
    unsigned short* bStT[2];
    if constexpr (OPB == 0) {
        bBase = Bp + (size_t)((t >> 4) + kBeg) * ldb + n0 + ((t & 15) << 3);
        #pragma unroll
        for (int e = 0; e < 2; ++e)
            bStV[e] = (uint4*)&Bs[(t >> 4) + e * 16][(t & 15) << 3];
    } else {
        bBase = Bp + (size_t)(n0 + (t >> 2)) * ldb + ((t & 3) << 3) + kBeg;
        #pragma unroll
        for (int e = 0; e < 2; ++e)
            bStT[e] = &Bs[(t & 3) << 3][(t >> 2) + e * 64];
    }

    v8f acc[2][4];
    #pragma unroll
    for (int mi = 0; mi < 2; ++mi)
        #pragma unroll
        for (int ni = 0; ni < 4; ++ni)
            acc[mi][ni] = (v8f){0.f,0.f,0.f,0.f,0.f,0.f,0.f,0.f};

    for (int k0 = 0; k0 < kSpan; k0 += BK) {
        uint4 va[2], vb[2];
        #pragma unroll
        for (int e = 0; e < 2; ++e) {
            if constexpr (OPA == 0)
                va[e] = *(const uint4*)(aBase + (size_t)e * 64 * lda);
            else
                va[e] = *(const uint4*)(aBase + (size_t)e * 16 * lda);
        }
        #pragma unroll
        for (int e = 0; e < 2; ++e) {
            if constexpr (OPB == 0)
                vb[e] = *(const uint4*)(bBase + (size_t)e * 16 * ldb);
            else
                vb[e] = *(const uint4*)(bBase + (size_t)e * 64 * ldb);
        }
        if constexpr (OPA == 0) aBase += BK; else aBase += (size_t)BK * lda;
        if constexpr (OPB == 0) bBase += (size_t)BK * ldb; else bBase += BK;

        __syncthreads();
        if constexpr (OPA == 0) {
            #pragma unroll
            for (int e = 0; e < 2; ++e) *aStV[e] = va[e];
        } else {
            #pragma unroll
            for (int e = 0; e < 2; ++e) scatter8(aStT[e], va[e], ASTR2);
        }
        if constexpr (OPB == 0) {
            #pragma unroll
            for (int e = 0; e < 2; ++e) *bStV[e] = vb[e];
        } else {
            #pragma unroll
            for (int e = 0; e < 2; ++e) scatter8(bStT[e], vb[e], BSTR2);
        }
        __syncthreads();

        // A 16x32 bf16 fragment (ISA 7.12.2)
        v16bf af[2];
        #pragma unroll
        for (int mi = 0; mi < 2; ++mi) {
            const int row = wr * 32 + mi * 16 + l16;
            v16bf a;
            #pragma unroll
            for (int v = 0; v < 8; ++v) {
                int kb = (v < 4) ? (8 * half + 2 * v)
                                 : (16 + 8 * half + 2 * (v - 4));
                a[2 * v]     = bfu(As[row][kb]);
                a[2 * v + 1] = bfu(As[row][kb + 1]);
            }
            af[mi] = a;
        }
        // B 32x16 bf16 fragment: lane%16 = N column, K in same order as A
        v16bf bfrag[4];
        #pragma unroll
        for (int ni = 0; ni < 4; ++ni) {
            const int col = wc * 64 + ni * 16;
            v16bf b;
            #pragma unroll
            for (int v = 0; v < 8; ++v) {
                int kb = (v < 4) ? (8 * half + 2 * v)
                                 : (16 + 8 * half + 2 * (v - 4));
                b[2 * v]     = bfu(Bs[kb][col + l16]);
                b[2 * v + 1] = bfu(Bs[kb + 1][col + l16]);
            }
            bfrag[ni] = b;
        }

        #pragma unroll
        for (int mi = 0; mi < 2; ++mi)
            #pragma unroll
            for (int ni = 0; ni < 4; ++ni)
                acc[mi][ni] = __builtin_amdgcn_wmma_f32_16x16x32_bf16(
                    false, af[mi], false, bfrag[ni],
                    (short)0, acc[mi][ni], false, false);
    }

    // ---- store partial C: uniform base + 32-bit running offsets ----
    float* Cp = C + (size_t)blockIdx.z * psz;
    #pragma unroll
    for (int mi = 0; mi < 2; ++mi)
        #pragma unroll
        for (int ni = 0; ni < 4; ++ni) {
            const int rbase = m0 + wr * 32 + mi * 16 + half * 8;
            const int cbase = n0 + wc * 64 + ni * 16 + l16;
            unsigned off = (unsigned)(rbase * ldc + cbase);
            #pragma unroll
            for (int r = 0; r < 8; ++r) {
                Cp[off] = acc[mi][ni][r];
                off += (unsigned)ldc;
            }
        }
}

// ===========================================================================
// GEMM #2: fp32 operands with in-kernel bf16 conversion (fallback when the
// workspace is too small for bf16 operand copies).
// ===========================================================================
template <int OPA, int OPB, int ACC>
__global__ __launch_bounds__(256)
void gemm_wmma_f(float* __restrict__ C, const float* __restrict__ A,
                 const float* __restrict__ Bp, int K, int lda, int ldb, int ldc)
{
    __shared__ unsigned short As[BM][ASTR];
    __shared__ unsigned short Bs[BK][BSTR];

    const int t    = threadIdx.x;
    const int lane = t & 31;
    const int w    = t >> 5;
    const int wr   = w >> 1;
    const int wc   = w & 1;
    const int half = lane >> 4;
    const int l16  = lane & 15;

    const int m0 = blockIdx.y * BM;
    const int n0 = blockIdx.x * BN;

    const float* aBase;
    unsigned*       aStP[4];
    unsigned short* aStT[4];
    if constexpr (OPA == 0) {
        aBase = A + (size_t)(m0 + (t >> 3)) * lda + ((t & 7) << 2);
        #pragma unroll
        for (int e = 0; e < 4; ++e)
            aStP[e] = (unsigned*)&As[(t >> 3) + e * 32][(t & 7) << 2];
    } else {
        aBase = A + (size_t)(t >> 5) * lda + m0 + (lane << 2);
        #pragma unroll
        for (int e = 0; e < 4; ++e)
            aStT[e] = &As[lane << 2][(t >> 5) + e * 8];
    }
    const float* bBase;
    unsigned*       bStP[4];
    unsigned short* bStT[4];
    if constexpr (OPB == 0) {
        bBase = Bp + (size_t)(t >> 5) * ldb + n0 + (lane << 2);
        #pragma unroll
        for (int e = 0; e < 4; ++e)
            bStP[e] = (unsigned*)&Bs[(t >> 5) + e * 8][lane << 2];
    } else {
        bBase = Bp + (size_t)(n0 + (t >> 3)) * ldb + ((t & 7) << 2);
        #pragma unroll
        for (int e = 0; e < 4; ++e)
            bStT[e] = &Bs[(t & 7) << 2][(t >> 3) + e * 32];
    }

    v8f acc[2][4];
    #pragma unroll
    for (int mi = 0; mi < 2; ++mi)
        #pragma unroll
        for (int ni = 0; ni < 4; ++ni) {
            if constexpr (ACC) {
                const int rbase = m0 + wr * 32 + mi * 16 + half * 8;
                const int cbase = n0 + wc * 64 + ni * 16 + l16;
                unsigned off = (unsigned)(rbase * ldc + cbase);
                v8f v;
                #pragma unroll
                for (int r = 0; r < 8; ++r) { v[r] = C[off]; off += (unsigned)ldc; }
                acc[mi][ni] = v;
            } else {
                acc[mi][ni] = (v8f){0.f,0.f,0.f,0.f,0.f,0.f,0.f,0.f};
            }
        }

    for (int k0 = 0; k0 < K; k0 += BK) {
        float4 va[4], vb[4];
        #pragma unroll
        for (int e = 0; e < 4; ++e) {
            if constexpr (OPA == 0)
                va[e] = *(const float4*)(aBase + (size_t)e * 32 * lda);
            else
                va[e] = *(const float4*)(aBase + (size_t)e * 8 * lda);
        }
        #pragma unroll
        for (int e = 0; e < 4; ++e) {
            if constexpr (OPB == 0)
                vb[e] = *(const float4*)(bBase + (size_t)e * 8 * ldb);
            else
                vb[e] = *(const float4*)(bBase + (size_t)e * 32 * ldb);
        }
        if constexpr (OPA == 0) aBase += BK; else aBase += (size_t)BK * lda;
        if constexpr (OPB == 0) bBase += (size_t)BK * ldb; else bBase += BK;

        __syncthreads();
        if constexpr (OPA == 0) {
            #pragma unroll
            for (int e = 0; e < 4; ++e) {
                aStP[e][0] = pk_bf16(va[e].x, va[e].y);
                aStP[e][1] = pk_bf16(va[e].z, va[e].w);
            }
        } else {
            #pragma unroll
            for (int e = 0; e < 4; ++e) {
                unsigned short* p = aStT[e];
                p[0]        = bf16r(va[e].x);
                p[ASTR]     = bf16r(va[e].y);
                p[2 * ASTR] = bf16r(va[e].z);
                p[3 * ASTR] = bf16r(va[e].w);
            }
        }
        if constexpr (OPB == 0) {
            #pragma unroll
            for (int e = 0; e < 4; ++e) {
                bStP[e][0] = pk_bf16(vb[e].x, vb[e].y);
                bStP[e][1] = pk_bf16(vb[e].z, vb[e].w);
            }
        } else {
            #pragma unroll
            for (int e = 0; e < 4; ++e) {
                unsigned short* p = bStT[e];
                p[0]        = bf16r(vb[e].x);
                p[BSTR]     = bf16r(vb[e].y);
                p[2 * BSTR] = bf16r(vb[e].z);
                p[3 * BSTR] = bf16r(vb[e].w);
            }
        }
        __syncthreads();

        v16bf af[2];
        #pragma unroll
        for (int mi = 0; mi < 2; ++mi) {
            const int row = wr * 32 + mi * 16 + l16;
            v16bf a;
            #pragma unroll
            for (int v = 0; v < 8; ++v) {
                int kb = (v < 4) ? (8 * half + 2 * v)
                                 : (16 + 8 * half + 2 * (v - 4));
                a[2 * v]     = bfu(As[row][kb]);
                a[2 * v + 1] = bfu(As[row][kb + 1]);
            }
            af[mi] = a;
        }
        v16bf bfrag[4];
        #pragma unroll
        for (int ni = 0; ni < 4; ++ni) {
            const int col = wc * 64 + ni * 16;
            v16bf b;
            #pragma unroll
            for (int v = 0; v < 8; ++v) {
                int kb = (v < 4) ? (8 * half + 2 * v)
                                 : (16 + 8 * half + 2 * (v - 4));
                b[2 * v]     = bfu(Bs[kb][col + l16]);
                b[2 * v + 1] = bfu(Bs[kb + 1][col + l16]);
            }
            bfrag[ni] = b;
        }

        #pragma unroll
        for (int mi = 0; mi < 2; ++mi)
            #pragma unroll
            for (int ni = 0; ni < 4; ++ni)
                acc[mi][ni] = __builtin_amdgcn_wmma_f32_16x16x32_bf16(
                    false, af[mi], false, bfrag[ni],
                    (short)0, acc[mi][ni], false, false);
    }

    #pragma unroll
    for (int mi = 0; mi < 2; ++mi)
        #pragma unroll
        for (int ni = 0; ni < 4; ++ni) {
            const int rbase = m0 + wr * 32 + mi * 16 + half * 8;
            const int cbase = n0 + wc * 64 + ni * 16 + l16;
            unsigned off = (unsigned)(rbase * ldc + cbase);
            #pragma unroll
            for (int r = 0; r < 8; ++r) {
                C[off] = acc[mi][ni][r];
                off += (unsigned)ldc;
            }
        }
}

// ---------------------------------------------------------------------------
// Elementwise helpers
// ---------------------------------------------------------------------------
__global__ void cvt_kernel(const float* __restrict__ in,
                           unsigned short* __restrict__ out, size_t n4)
{
    const size_t stride = (size_t)gridDim.x * 256;
    for (size_t i = (size_t)blockIdx.x * 256 + threadIdx.x; i < n4; i += stride) {
        float4 v = ((const float4*)in)[i];
        uint2 p;
        p.x = pk_bf16(v.x, v.y);
        p.y = pk_bf16(v.z, v.w);
        ((uint2*)out)[i] = p;
    }
}

__global__ void rowsum_kernel(const float* __restrict__ A, float* __restrict__ out,
                              int cols)
{
    __shared__ float red[256];
    const int r = blockIdx.x;
    float s = 0.f;
    for (int c = threadIdx.x; c < cols; c += 256)
        s += A[(size_t)r * cols + c];
    red[threadIdx.x] = s;
    __syncthreads();
    for (int off = 128; off > 0; off >>= 1) {
        if ((int)threadIdx.x < off) red[threadIdx.x] += red[threadIdx.x + off];
        __syncthreads();
    }
    if (threadIdx.x == 0) out[r] = red[0];
}

__global__ void colsum_kernel(const float* __restrict__ A, float* __restrict__ out,
                              int rows, int cols)
{
    const int c = blockIdx.x * 256 + threadIdx.x;
    if (c >= cols) return;
    float s = 0.f;
    for (int r = 0; r < rows; ++r) s += A[(size_t)r * cols + c];
    out[c] = s;
}

// Reduce S split-K partials + epilogue:
//   v = sum_s parts[s*psz + i]; v = v*inv(deg[r]) + bias[c]; relu;
//   optionally write f32 and/or bf16. One row (256 cols) per block.
__global__ void reduce_ep_kernel(const float* __restrict__ parts, int S, int psz,
                                 float* __restrict__ outF,
                                 unsigned short* __restrict__ outH,
                                 const float* __restrict__ deg,
                                 const float* __restrict__ bias, int relu)
{
    const int r = blockIdx.x, c = threadIdx.x;
    const int i = r * HIDC + c;
    float v = 0.f;
    for (int s = 0; s < S; ++s) v += parts[(size_t)s * psz + i];
    float sc = 1.0f;
    if (deg) { const float d = deg[r]; sc = d > 0.f ? 1.0f / d : 0.0f; }
    v = v * sc + (bias ? bias[c] : 0.0f);
    if (relu) v = fmaxf(v, 0.0f);
    if (outF) outF[i] = v;
    if (outH) outH[i] = bf16r(v);
}

// in-place f32 scale/bias (fallback path)
__global__ void scale_bias_kernel(float* __restrict__ X, const float* __restrict__ deg,
                                  const float* __restrict__ bias, int relu)
{
    const int r = blockIdx.x, c = threadIdx.x;
    const size_t i = (size_t)r * HIDC + c;
    const float d = deg[r];
    const float s = d > 0.f ? 1.0f / d : 0.0f;
    float v = X[i] * s + (bias ? bias[c] : 0.0f);
    if (relu) v = fmaxf(v, 0.0f);
    X[i] = v;
}

__global__ void combine_kernel(float* __restrict__ out, const float* __restrict__ m1,
                               const float* __restrict__ m2, const float* __restrict__ b)
{
    const size_t i = (size_t)blockIdx.x * HIDC + threadIdx.x;
    out[i] = 0.5f * (m1[i] + m2[i] + b[threadIdx.x]);
}

// ---------------------------------------------------------------------------
// Host-side orchestration
// ---------------------------------------------------------------------------
static void gemmH(hipStream_t s, float* Cparts, const unsigned short* A,
                  const unsigned short* B, int M, int K,
                  int lda, int ldb, int S, int opA, int opB)
{
    dim3 grid(HIDC / BN, M / BM, S);
    const int kSpan = K / S;
    const int psz = M * HIDC;
    if (opA == 0 && opB == 0)
        gemm_wmma_h<0,0><<<grid, 256, 0, s>>>(Cparts, A, B, kSpan, lda, ldb, HIDC, psz);
    else if (opA == 1)
        gemm_wmma_h<1,0><<<grid, 256, 0, s>>>(Cparts, A, B, kSpan, lda, ldb, HIDC, psz);
    else
        gemm_wmma_h<0,1><<<grid, 256, 0, s>>>(Cparts, A, B, kSpan, lda, ldb, HIDC, psz);
}

static void gemmF(hipStream_t s, float* C, const float* A, const float* B,
                  int M, int K, int lda, int ldb, int opA, int opB, int acc)
{
    dim3 grid(HIDC / BN, M / BM);
    if (opA == 0 && opB == 0 && acc == 0)
        gemm_wmma_f<0,0,0><<<grid, 256, 0, s>>>(C, A, B, K, lda, ldb, HIDC);
    else if (opA == 0 && opB == 0 && acc == 1)
        gemm_wmma_f<0,0,1><<<grid, 256, 0, s>>>(C, A, B, K, lda, ldb, HIDC);
    else if (opA == 1 && opB == 0)
        gemm_wmma_f<1,0,0><<<grid, 256, 0, s>>>(C, A, B, K, lda, ldb, HIDC);
    else if (opA == 0 && opB == 1 && acc == 0)
        gemm_wmma_f<0,1,0><<<grid, 256, 0, s>>>(C, A, B, K, lda, ldb, HIDC);
    else
        gemm_wmma_f<0,1,1><<<grid, 256, 0, s>>>(C, A, B, K, lda, ldb, HIDC);
}

// ---- bf16-operand branch with split-K (preferred) ----
static void run_branch_h(hipStream_t s,
                         const float* fea0, const float* fea1, const float* adj,
                         const float* W1, const float* b1,
                         const float* W2, const float* b2,
                         const float* Wl, const float* bl,
                         int Nn, float* out,
                         float* dr, float* dc, float* P1, float* P2, float* PP,
                         unsigned short* Q0, unsigned short* Q1,
                         unsigned short* W1B, unsigned short* WlB, unsigned short* W2B,
                         unsigned short* adjB, unsigned short* f0B, unsigned short* f1B)
{
    const int H = HIDC;
    const int S = 4;                       // split-K factor for big GEMMs
    const int psz = Nn * H;
    const size_t nn4 = (size_t)Nn * Nn / 4;
    float* PPh = PP + (size_t)S * psz;     // second partial group (concat GEMMs)

    cvt_kernel<<<2048, 256, 0, s>>>(adj,  adjB, nn4);
    cvt_kernel<<<2048, 256, 0, s>>>(fea0, f0B,  nn4);
    cvt_kernel<<<2048, 256, 0, s>>>(fea1, f1B,  nn4);
    cvt_kernel<<<256,  256, 0, s>>>(W1, W1B, (size_t)2 * Nn * H / 4);
    cvt_kernel<<<256,  256, 0, s>>>(Wl, WlB, (size_t)2 * Nn * H / 4);
    cvt_kernel<<<64,   256, 0, s>>>(W2, W2B, (size_t)H * H / 4);

    rowsum_kernel<<<Nn, 256, 0, s>>>(adj, dr, Nn);
    colsum_kernel<<<(Nn + 255) / 256, 256, 0, s>>>(adj, dc, Nn, Nn);

    // Q0 = bf16(fea @ W1)
    gemmH(s, PP,  f0B, W1B,                  Nn, Nn, Nn, H, S, 0, 0);
    gemmH(s, PPh, f1B, W1B + (size_t)Nn * H, Nn, Nn, Nn, H, S, 0, 0);
    reduce_ep_kernel<<<Nn, 256, 0, s>>>(PP, 2 * S, psz, nullptr, Q0, nullptr, nullptr, 0);

    // Q1 = bf16(Binv * (adj @ Q0))
    gemmH(s, PP, adjB, Q0, Nn, Nn, Nn, H, S, 0, 0);
    reduce_ep_kernel<<<Nn, 256, 0, s>>>(PP, S, psz, nullptr, Q1, dr, nullptr, 0);

    // Q0 = bf16(relu(Dinv * (adjT @ Q1) + b1))
    gemmH(s, PP, adjB, Q1, Nn, Nn, Nn, H, S, 1, 0);
    reduce_ep_kernel<<<Nn, 256, 0, s>>>(PP, S, psz, nullptr, Q0, dc, b1, 1);

    // Q1 = bf16(Q0 @ W2)
    gemmH(s, PP, Q0, W2B, Nn, H, H, H, 1, 0, 0);
    reduce_ep_kernel<<<Nn, 256, 0, s>>>(PP, 1, psz, nullptr, Q1, nullptr, nullptr, 0);

    // Q0 = bf16(Binv * (adj @ Q1))
    gemmH(s, PP, adjB, Q1, Nn, Nn, Nn, H, S, 0, 0);
    reduce_ep_kernel<<<Nn, 256, 0, s>>>(PP, S, psz, nullptr, Q0, dr, nullptr, 0);

    // P1 = m1 = Dinv * (adjT @ Q0) + b2   (fp32)
    gemmH(s, PP, adjB, Q0, Nn, Nn, Nn, H, S, 1, 0);
    reduce_ep_kernel<<<Nn, 256, 0, s>>>(PP, S, psz, P1, nullptr, dc, b2, 0);

    // P2 = m2 - bl = fea @ Wl^T
    gemmH(s, PP,  f0B, WlB,      Nn, Nn, Nn, 2 * Nn, S, 0, 1);
    gemmH(s, PPh, f1B, WlB + Nn, Nn, Nn, Nn, 2 * Nn, S, 0, 1);
    reduce_ep_kernel<<<Nn, 256, 0, s>>>(PP, 2 * S, psz, P2, nullptr, nullptr, nullptr, 0);

    combine_kernel<<<Nn, 256, 0, s>>>(out, P1, P2, bl);
}

// ---- fp32-operand fallback branch ----
static void run_branch_f(hipStream_t s,
                         const float* fea0, const float* fea1, const float* adj,
                         const float* W1, const float* b1,
                         const float* W2, const float* b2,
                         const float* Wl, const float* bl,
                         int Nn, float* out,
                         float* dr, float* dc, float* P0, float* P1, float* P2)
{
    const int H = HIDC;

    rowsum_kernel<<<Nn, 256, 0, s>>>(adj, dr, Nn);
    colsum_kernel<<<(Nn + 255) / 256, 256, 0, s>>>(adj, dc, Nn, Nn);

    gemmF(s, P0, fea0, W1,                  Nn, Nn, Nn, H, 0, 0, 0);
    gemmF(s, P0, fea1, W1 + (size_t)Nn * H, Nn, Nn, Nn, H, 0, 0, 1);

    gemmF(s, P1, adj, P0, Nn, Nn, Nn, H, 0, 0, 0);
    scale_bias_kernel<<<Nn, 256, 0, s>>>(P1, dr, nullptr, 0);

    gemmF(s, P0, adj, P1, Nn, Nn, Nn, H, 1, 0, 0);
    scale_bias_kernel<<<Nn, 256, 0, s>>>(P0, dc, b1, 1);

    gemmF(s, P1, P0, W2, Nn, H, H, H, 0, 0, 0);

    gemmF(s, P2, adj, P1, Nn, Nn, Nn, H, 0, 0, 0);
    scale_bias_kernel<<<Nn, 256, 0, s>>>(P2, dr, nullptr, 0);

    gemmF(s, P1, adj, P2, Nn, Nn, Nn, H, 1, 0, 0);
    scale_bias_kernel<<<Nn, 256, 0, s>>>(P1, dc, b2, 0);

    gemmF(s, P2, fea0, Wl,      Nn, Nn, Nn, 2 * Nn, 0, 1, 0);
    gemmF(s, P2, fea1, Wl + Nn, Nn, Nn, Nn, 2 * Nn, 0, 1, 1);

    combine_kernel<<<Nn, 256, 0, s>>>(out, P1, P2, bl);
}

extern "C" void kernel_launch(void* const* d_in, const int* in_sizes, int n_in,
                              void* d_out, int out_size, void* d_ws, size_t ws_size,
                              hipStream_t stream)
{
    (void)in_sizes; (void)n_in; (void)out_size;

    const float* m_f    = (const float*)d_in[0];
    const float* d_f    = (const float*)d_in[1];
    const float* c_func = (const float*)d_in[2];
    const float* c_gs   = (const float*)d_in[3];
    const float* d_ss   = (const float*)d_in[4];
    const float* dd_g   = (const float*)d_in[5];
    const float* W1m    = (const float*)d_in[6];
    const float* b1m    = (const float*)d_in[7];
    const float* W2m    = (const float*)d_in[8];
    const float* b2m    = (const float*)d_in[9];
    const float* W1d    = (const float*)d_in[10];
    const float* b1d    = (const float*)d_in[11];
    const float* W2d    = (const float*)d_in[12];
    const float* b2d    = (const float*)d_in[13];
    const float* Wm     = (const float*)d_in[14];
    const float* bm     = (const float*)d_in[15];
    const float* Wd     = (const float*)d_in[16];
    const float* bd     = (const float*)d_in[17];

    float* out = (float*)d_out;

    const size_t PSZ = (size_t)6144 * HIDC;      // 1,572,864 elems
    const size_t ADJ = (size_t)6144 * 6144;      // 37,748,736 elems

    // fp32 region
    float* dr = (float*)d_ws;
    float* dc = dr + 8192;
    float* P0 = dr + 16384;
    float* P1 = P0 + PSZ;
    float* P2 = P1 + PSZ;
    float* PP = P2 + PSZ;                        // 8 split-K partial buffers
    // bf16 region
    unsigned short* Q0   = (unsigned short*)(PP + 8 * PSZ);
    unsigned short* Q1   = Q0 + PSZ;
    unsigned short* W1B  = Q1 + PSZ;
    unsigned short* WlB  = W1B + (size_t)12288 * HIDC;
    unsigned short* W2B  = WlB + (size_t)12288 * HIDC;
    unsigned short* adjB = W2B + (size_t)HIDC * HIDC;
    unsigned short* f0B  = adjB + ADJ;
    unsigned short* f1B  = f0B + ADJ;

    const size_t needF = (16384 + 3 * PSZ) * sizeof(float);
    const size_t needH = (16384 + 11 * PSZ) * sizeof(float) +
        (2 * PSZ + 2 * (size_t)12288 * HIDC + (size_t)HIDC * HIDC + 3 * ADJ)
            * sizeof(unsigned short);

    if (ws_size >= needH) {
        run_branch_h(stream, c_func, c_gs, m_f, W1m, b1m, W2m, b2m, Wm, bm,
                     6144, out, dr, dc, P1, P2, PP,
                     Q0, Q1, W1B, WlB, W2B, adjB, f0B, f1B);
        run_branch_h(stream, d_ss, dd_g, d_f, W1d, b1d, W2d, b2d, Wd, bd,
                     4096, out + (size_t)6144 * HIDC, dr, dc, P1, P2, PP,
                     Q0, Q1, W1B, WlB, W2B, adjB, f0B, f1B);
    } else {
        run_branch_f(stream, c_func, c_gs, m_f, W1m, b1m, W2m, b2m, Wm, bm,
                     6144, out, dr, dc, P0, P1, P2);
        run_branch_f(stream, d_ss, dd_g, d_f, W1d, b1d, W2d, b2d, Wd, bd,
                     4096, out + (size_t)6144 * HIDC, dr, dc, P0, P1, P2);
    }
}
